// Attention_27565100106273
// MI455X (gfx1250) — matrix operand order repaired and checked by reading, not compiled, not run on hardware
//
#include <hip/hip_runtime.h>
#include <hip/hip_bf16.h>

// ---------------------------------------------------------------------------
// Attention block for MI455X (gfx1250, wave32, WMMA + TDM).
// fp32 in/out, bf16 WMMA (v_wmma_f32_16x16x32_bf16) with fp32 accumulation.
// Row-major LDS tile staging via Tensor Data Mover where possible.
// ---------------------------------------------------------------------------

#define DIM      1024
#define NHEADS   16
#define HDIM     64
#define SEQ      2048
#define BATCH    2
#define MROWS    (BATCH * SEQ)     // 4096
#define QKVCOLS  (3 * DIM)         // 3072
#define SCALE    0.125f            // HDIM^-0.5

typedef __attribute__((ext_vector_type(16))) __bf16 v16bf;
typedef __attribute__((ext_vector_type(8)))  float  v8f;

#if defined(__gfx1250__) && __has_builtin(__builtin_amdgcn_tensor_load_to_lds)
#define USE_TDM 1
#else
#define USE_TDM 0
#endif

union BFrag {
    v16bf  v;
    __bf16 h[16];
};

union LdU {
    uint4  q[2];
    __bf16 h[16];
};

__device__ __forceinline__ v8f zero8() {
    v8f z;
#pragma unroll
    for (int i = 0; i < 8; ++i) z[i] = 0.0f;
    return z;
}

// A-fragment (16x32, 16-bit): lane half = lane>>4, row m = lane&15.
// VGPR j holds K = {2j, 2j+1} (+16 if half==1).
__device__ __forceinline__ v16bf load_a_frag(const __bf16* __restrict__ base,
                                             int ld, int lane) {
    const int half = (lane >> 4) & 1;
    const int m    = lane & 15;
    const __bf16* row = base + m * ld;
    BFrag f;
#pragma unroll
    for (int j = 0; j < 8; ++j) {
        const int kb = 2 * j + half * 16;
        f.h[2 * j]     = row[kb];
        f.h[2 * j + 1] = row[kb + 1];
    }
    return f.v;
}

// B-fragment (32x16, 16-bit) from LDS stored n-major: element (k,n) at
// base[n*ld + k]. Lane: n = lane&15, K = 16*half + e (16 contiguous).
__device__ __forceinline__ v16bf load_b_frag(const __bf16* __restrict__ base,
                                             int ld, int lane) {
    const int half = (lane >> 4) & 1;
    const int n    = lane & 15;
    const __bf16* col = base + n * ld + half * 16;
    BFrag f;
#pragma unroll
    for (int e = 0; e < 16; ++e) f.h[e] = col[e];
    return f.v;
}

__device__ __forceinline__ v8f wmma_bf16(v16bf a, v16bf b, v8f c) {
    return __builtin_amdgcn_wmma_f32_16x16x32_bf16(
        /*neg_a=*/false, a, /*neg_b=*/false, b,
        /*c_mod=*/(short)0, c, /*reuse_a=*/false, /*reuse_b=*/false);
}

#if USE_TDM
typedef __attribute__((ext_vector_type(4))) unsigned int u32x4;
typedef __attribute__((ext_vector_type(8))) int          i32x8;
typedef __attribute__((ext_vector_type(4))) int          i32x4;

// TDM 2D tile load, bf16 elements, optional LDS row padding.
//   tensor_d0/d1 : tensor extents (elements) relative to tile origin
//   tile_d0/d1   : tile extents (elements); d0 must be 4B-multiple for pad
//   stride0      : tensor dim0 stride in elements
//   pad_iv_code  : pad interval (0=2,1=4,2=8,3=16,4=32,... DWORDs)
//   pad_amt_code : pad amount (code+1 DWORDs)
__device__ __forceinline__ void tdm_load_2d_bf16(
    unsigned lds_off, const void* gptr,
    unsigned tensor_d0, unsigned tensor_d1,
    unsigned tile_d0, unsigned tile_d1,
    unsigned long long stride0,
    unsigned pad_iv_code, unsigned pad_amt_code, bool pad_en) {
    const unsigned long long ga = (unsigned long long)(size_t)gptr;
    u32x4 g0;
    g0[0] = 1u;                                           // count=1, user D#
    g0[1] = lds_off;                                      // LDS byte address
    g0[2] = (unsigned)(ga & 0xFFFFFFFFu);                 // global addr lo
    g0[3] = (unsigned)((ga >> 32) & 0x1FFFFFFu) | (2u << 30);  // hi | type=2

    unsigned w0 = (1u << 16);                             // data_size = 2B
    if (pad_en)
        w0 |= (1u << 20) | (pad_iv_code << 22) | (pad_amt_code << 25);
    i32x8 g1;
    g1[0] = (int)w0;
    g1[1] = (int)((tensor_d0 & 0xFFFFu) << 16);           // dim0[15:0]
    g1[2] = (int)(((tensor_d0 >> 16) & 0xFFFFu) |
                  ((tensor_d1 & 0xFFFFu) << 16));         // dim0[31:16]|dim1 lo
    g1[3] = (int)(((tensor_d1 >> 16) & 0xFFFFu) |
                  ((tile_d0 & 0xFFFFu) << 16));           // dim1 hi | tile0
    g1[4] = (int)(tile_d1 & 0xFFFFu);                     // tile1; tile2=0
    g1[5] = (int)(unsigned)(stride0 & 0xFFFFFFFFull);     // stride0 lo
    g1[6] = (int)(unsigned)((stride0 >> 32) & 0xFFFFull); // stride0 hi
    g1[7] = 0;
    const i32x4 z4 = {0, 0, 0, 0};
    const i32x8 z8 = {0, 0, 0, 0, 0, 0, 0, 0};
    // 6-arg form (clang-23 / therock-10.0): groups 2/3 unused for 2D tiles.
    __builtin_amdgcn_tensor_load_to_lds(g0, g1, z4, z4, z8, 0);
}

__device__ __forceinline__ unsigned lds_offset(const void* p) {
    return (unsigned)(size_t)p;   // LDS aperture: low 32 bits = byte offset
}
#endif

// ---------------------------------------------------------------------------
// fp32 -> bf16 conversion (grid-stride)
// ---------------------------------------------------------------------------
__global__ void cvt_f32_bf16(const float* __restrict__ in,
                             __bf16* __restrict__ out, int n) {
    int i = blockIdx.x * blockDim.x + threadIdx.x;
    const int stride = gridDim.x * blockDim.x;
    for (; i < n; i += stride) out[i] = (__bf16)in[i];
}

// ---------------------------------------------------------------------------
// QKV GEMM: X[4096x1024] * Wqkv[1024x3072] -> Q/K/V in [B,H,N,Dh] bf16.
// Block: 128 threads (4 waves). Tile: 128(M) x 64(N), K-chunks of 32.
// Each wave owns two 16-row A-fragments -> 8 WMMAs per chunk.
// ---------------------------------------------------------------------------
__global__ __launch_bounds__(128)
void qkv_gemm(const __bf16* __restrict__ X, const __bf16* __restrict__ W,
              __bf16* __restrict__ Q, __bf16* __restrict__ K,
              __bf16* __restrict__ V) {
    __shared__ alignas(16) __bf16 lA[128 * 40];  // row-major [m][k], ld=40
    __shared__ alignas(16) __bf16 lB[64 * 40];   // n-major  [n][k], ld=40

    const int tid  = threadIdx.x;
    const int lane = tid & 31;
    const int wave = tid >> 5;
    const int half = (lane >> 4) & 1;
    const int bm   = blockIdx.x * 128;
    const int bn   = blockIdx.y * 64;

    v8f acc[2][4];
#pragma unroll
    for (int mt = 0; mt < 2; ++mt)
#pragma unroll
        for (int nt = 0; nt < 4; ++nt) acc[mt][nt] = zero8();

    for (int kc = 0; kc < DIM; kc += 32) {
        __syncthreads();
#if USE_TDM
        if (wave == 0)   // A tile 128x32 via TDM (pad 16 DW -> +4 DW => ld=40)
            tdm_load_2d_bf16(lds_offset(lA), X + (size_t)bm * DIM + kc,
                             32, 128, 32, 128, DIM, 3, 3, true);
#else
        {   // A tile: 128 rows x 32 k. Each thread: one row (32 bf16).
            const uint4* src = reinterpret_cast<const uint4*>(
                X + (size_t)(bm + tid) * DIM + kc);
            uint4 d0 = src[0], d1 = src[1], d2 = src[2], d3 = src[3];
            uint4* dst = reinterpret_cast<uint4*>(&lA[tid * 40]);
            dst[0] = d0; dst[1] = d1; dst[2] = d2; dst[3] = d3;
        }
#endif
        {   // B tile: 32 k x 64 n, stored transposed into lB[n][k].
            const int k = tid >> 2, np = (tid & 3) * 16;
            LdU u;
            const uint4* src = reinterpret_cast<const uint4*>(
                W + (size_t)(kc + k) * QKVCOLS + bn + np);
            u.q[0] = src[0]; u.q[1] = src[1];
#pragma unroll
            for (int e = 0; e < 16; ++e) lB[(np + e) * 40 + k] = u.h[e];
        }
#if USE_TDM
        if (wave == 0) __builtin_amdgcn_s_wait_tensorcnt(0);
#endif
        __syncthreads();

        const v16bf a0 = load_a_frag(&lA[(wave * 32) * 40], 40, lane);
        const v16bf a1 = load_a_frag(&lA[(wave * 32 + 16) * 40], 40, lane);
#pragma unroll
        for (int nt = 0; nt < 4; ++nt) {
            v16bf b = load_b_frag(&lB[(nt * 16) * 40], 40, lane);
            acc[0][nt] = wmma_bf16(a0, b, acc[0][nt]);
            acc[1][nt] = wmma_bf16(a1, b, acc[1][nt]);
        }
    }

    // Store: split columns into Q/K/V with [B,H,N,Dh] layout.
#pragma unroll
    for (int nt = 0; nt < 4; ++nt) {
        const int col = bn + nt * 16 + (lane & 15);
        __bf16* dst = (col < DIM) ? Q : ((col < 2 * DIM) ? K : V);
        const int c2 = col & (DIM - 1);
        const int h = c2 >> 6, d = c2 & 63;
#pragma unroll
        for (int mt = 0; mt < 2; ++mt)
#pragma unroll
            for (int r = 0; r < 8; ++r) {
                const int mg = bm + wave * 32 + mt * 16 + r + 8 * half;
                const int b = mg >> 11, n = mg & (SEQ - 1);
                dst[(((size_t)(b * NHEADS + h) * SEQ) + n) * HDIM + d] =
                    (__bf16)acc[mt][nt][r];
            }
    }
}

// ---------------------------------------------------------------------------
// Flash attention: grid (B*H, N/64). 4 waves/block; wave owns 16 queries.
// Key chunks of 32; S via 4 WMMAs, online softmax, O += P*V via 4 WMMAs.
// Q tile and K chunks staged via TDM; V transposed manually.
// ---------------------------------------------------------------------------
__global__ __launch_bounds__(128)
void attn_kernel(const __bf16* __restrict__ Q, const __bf16* __restrict__ K,
                 const __bf16* __restrict__ V, __bf16* __restrict__ O) {
    __shared__ alignas(16) __bf16 lQ[64 * 72];       // [query][feat], ld=72
    __shared__ alignas(16) __bf16 lK[32 * 72];       // [key][feat],   ld=72
    __shared__ alignas(16) __bf16 lVt[64 * 40];      // [feat][key],   ld=40
    __shared__ alignas(16) __bf16 lP[4][16 * 40];    // per-wave [m][key], ld=40

    const int tid   = threadIdx.x;
    const int lane  = tid & 31;
    const int wave  = tid >> 5;
    const int half  = (lane >> 4) & 1;
    const int bh    = blockIdx.x;            // b*NHEADS + h
    const int qbase = blockIdx.y * 64;

    const __bf16* Qp = Q + (size_t)bh * SEQ * HDIM;
    const __bf16* Kp = K + (size_t)bh * SEQ * HDIM;
    const __bf16* Vp = V + (size_t)bh * SEQ * HDIM;

#if USE_TDM
    if (wave == 0) {     // Q tile 64x64 (pad 32 DW -> +4 DW => ld=72)
        tdm_load_2d_bf16(lds_offset(lQ), Qp + (size_t)qbase * HDIM,
                         64, 64, 64, 64, HDIM, 4, 3, true);
        __builtin_amdgcn_s_wait_tensorcnt(0);
    }
#else
    {   // Load Q tile 64x64 once: each thread one half-row (32 bf16).
        const int row = tid >> 1, part = (tid & 1) * 32;
        const uint4* src = reinterpret_cast<const uint4*>(
            Qp + (size_t)(qbase + row) * HDIM + part);
        uint4 a0 = src[0], a1 = src[1], a2 = src[2], a3 = src[3];
        uint4* dst = reinterpret_cast<uint4*>(&lQ[row * 72 + part]);
        dst[0] = a0; dst[1] = a1; dst[2] = a2; dst[3] = a3;
    }
#endif
    __syncthreads();

    const v16bf qa0 = load_a_frag(&lQ[(wave * 16) * 72], 72, lane);       // k 0..31
    const v16bf qa1 = load_a_frag(&lQ[(wave * 16) * 72 + 32], 72, lane);  // k 32..63

    float mrow[8], lrow[8];
    v8f o[4];
#pragma unroll
    for (int r = 0; r < 8; ++r) { mrow[r] = -1e30f; lrow[r] = 0.0f; }
#pragma unroll
    for (int ft = 0; ft < 4; ++ft) o[ft] = zero8();

    for (int kt = 0; kt < SEQ; kt += 32) {
        __syncthreads();
#if USE_TDM
        if (wave == 0)   // K chunk 32x64 row-major via TDM
            tdm_load_2d_bf16(lds_offset(lK), Kp + (size_t)kt * HDIM,
                             64, 32, 64, 32, HDIM, 4, 3, true);
#endif
        {   // Stage V transposed (and K manually when no TDM).
            const int key = tid >> 2, part = (tid & 3) * 16;
#if !USE_TDM
            const uint4* ks = reinterpret_cast<const uint4*>(
                Kp + (size_t)(kt + key) * HDIM + part);
            uint4 k0 = ks[0], k1 = ks[1];
            uint4* kd = reinterpret_cast<uint4*>(&lK[key * 72 + part]);
            kd[0] = k0; kd[1] = k1;
#endif
            LdU vv;
            const uint4* vs = reinterpret_cast<const uint4*>(
                Vp + (size_t)(kt + key) * HDIM + part);
            vv.q[0] = vs[0]; vv.q[1] = vs[1];
#pragma unroll
            for (int e = 0; e < 16; ++e) lVt[(part + e) * 40 + key] = vv.h[e];
        }
#if USE_TDM
        if (wave == 0) __builtin_amdgcn_s_wait_tensorcnt(0);
#endif
        __syncthreads();

        // S = Q * K^T for 32 keys: two 16x16 C-tiles, each accumulated over Dh.
        v8f s0 = zero8(), s1 = zero8();
        s0 = wmma_bf16(qa0, load_b_frag(&lK[0], 72, lane), s0);
        s0 = wmma_bf16(qa1, load_b_frag(&lK[32], 72, lane), s0);
        s1 = wmma_bf16(qa0, load_b_frag(&lK[16 * 72], 72, lane), s1);
        s1 = wmma_bf16(qa1, load_b_frag(&lK[16 * 72 + 32], 72, lane), s1);

        // Online softmax. C layout: row m = r + 8*half, col n = lane&15.
#pragma unroll
        for (int r = 0; r < 8; ++r) {
            float a = s0[r] * SCALE;
            float b = s1[r] * SCALE;
            float mloc = fmaxf(a, b);
#pragma unroll
            for (int off = 1; off < 16; off <<= 1)
                mloc = fmaxf(mloc, __shfl_xor(mloc, off, 32));
            const float mnew = fmaxf(mrow[r], mloc);
            const float corr = __expf(mrow[r] - mnew);
            const float p0 = __expf(a - mnew);
            const float p1 = __expf(b - mnew);
            float rsum = p0 + p1;
#pragma unroll
            for (int off = 1; off < 16; off <<= 1)
                rsum += __shfl_xor(rsum, off, 32);
            lrow[r] = lrow[r] * corr + rsum;
            mrow[r] = mnew;
#pragma unroll
            for (int ft = 0; ft < 4; ++ft) o[ft][r] *= corr;
            // P -> LDS (own wave region; same-wave LDS ops are in-order).
            lP[wave][(r + 8 * half) * 40 + (lane & 15)]      = (__bf16)p0;
            lP[wave][(r + 8 * half) * 40 + 16 + (lane & 15)] = (__bf16)p1;
        }

        // O += P * V  (P: 16x32 A-fragment; V^T staged n-major).
        const v16bf pa = load_a_frag(&lP[wave][0], 40, lane);
#pragma unroll
        for (int ft = 0; ft < 4; ++ft) {
            v16bf vb = load_b_frag(&lVt[(ft * 16) * 40], 40, lane);
            o[ft] = wmma_bf16(pa, vb, o[ft]);
        }
    }

    // Epilogue: normalize and store to Ob in [B,N,C] bf16 layout.
    const int b = bh >> 4, h = bh & 15;
#pragma unroll
    for (int ft = 0; ft < 4; ++ft) {
        const int col = h * HDIM + ft * 16 + (lane & 15);
#pragma unroll
        for (int r = 0; r < 8; ++r) {
            const int ng = qbase + wave * 16 + r + 8 * half;
            const float val = o[ft][r] / lrow[r];
            O[((size_t)(b * SEQ + ng)) * DIM + col] = (__bf16)val;
        }
    }
}

// ---------------------------------------------------------------------------
// Output projection: Ob[4096x1024] * Wp[1024x1024] + bias -> fp32 out.
// Same 128x64 tiling as qkv_gemm.
// ---------------------------------------------------------------------------
__global__ __launch_bounds__(128)
void proj_gemm(const __bf16* __restrict__ A, const __bf16* __restrict__ W,
               const float* __restrict__ bias, float* __restrict__ out) {
    __shared__ alignas(16) __bf16 lA[128 * 40];
    __shared__ alignas(16) __bf16 lB[64 * 40];

    const int tid  = threadIdx.x;
    const int lane = tid & 31;
    const int wave = tid >> 5;
    const int half = (lane >> 4) & 1;
    const int bm   = blockIdx.x * 128;
    const int bn   = blockIdx.y * 64;

    v8f acc[2][4];
#pragma unroll
    for (int mt = 0; mt < 2; ++mt)
#pragma unroll
        for (int nt = 0; nt < 4; ++nt) acc[mt][nt] = zero8();

    for (int kc = 0; kc < DIM; kc += 32) {
        __syncthreads();
#if USE_TDM
        if (wave == 0)
            tdm_load_2d_bf16(lds_offset(lA), A + (size_t)bm * DIM + kc,
                             32, 128, 32, 128, DIM, 3, 3, true);
#else
        {
            const uint4* src = reinterpret_cast<const uint4*>(
                A + (size_t)(bm + tid) * DIM + kc);
            uint4 d0 = src[0], d1 = src[1], d2 = src[2], d3 = src[3];
            uint4* dst = reinterpret_cast<uint4*>(&lA[tid * 40]);
            dst[0] = d0; dst[1] = d1; dst[2] = d2; dst[3] = d3;
        }
#endif
        {
            const int k = tid >> 2, np = (tid & 3) * 16;
            LdU u;
            const uint4* src = reinterpret_cast<const uint4*>(
                W + (size_t)(kc + k) * DIM + bn + np);
            u.q[0] = src[0]; u.q[1] = src[1];
#pragma unroll
            for (int e = 0; e < 16; ++e) lB[(np + e) * 40 + k] = u.h[e];
        }
#if USE_TDM
        if (wave == 0) __builtin_amdgcn_s_wait_tensorcnt(0);
#endif
        __syncthreads();

        const v16bf a0 = load_a_frag(&lA[(wave * 32) * 40], 40, lane);
        const v16bf a1 = load_a_frag(&lA[(wave * 32 + 16) * 40], 40, lane);
#pragma unroll
        for (int nt = 0; nt < 4; ++nt) {
            v16bf b = load_b_frag(&lB[(nt * 16) * 40], 40, lane);
            acc[0][nt] = wmma_bf16(a0, b, acc[0][nt]);
            acc[1][nt] = wmma_bf16(a1, b, acc[1][nt]);
        }
    }

#pragma unroll
    for (int nt = 0; nt < 4; ++nt) {
        const int col = bn + nt * 16 + (lane & 15);
        const float bv = bias[col];
#pragma unroll
        for (int mt = 0; mt < 2; ++mt)
#pragma unroll
            for (int r = 0; r < 8; ++r) {
                const int mg = bm + wave * 32 + mt * 16 + r + 8 * half;
                out[(size_t)mg * DIM + col] = acc[mt][nt][r] + bv;
            }
    }
}

// ---------------------------------------------------------------------------
// Host launcher
// ---------------------------------------------------------------------------
extern "C" void kernel_launch(void* const* d_in, const int* in_sizes, int n_in,
                              void* d_out, int out_size, void* d_ws, size_t ws_size,
                              hipStream_t stream) {
    const float* x     = (const float*)d_in[0];
    const float* wqkv  = (const float*)d_in[1];
    const float* wproj = (const float*)d_in[2];
    const float* bproj = (const float*)d_in[3];
    float* out = (float*)d_out;

    char* ws = (char*)d_ws;
    const size_t nX   = (size_t)MROWS * DIM;          // 4M
    const size_t nWq  = (size_t)DIM * QKVCOLS;        // 3M
    const size_t nWp  = (size_t)DIM * DIM;            // 1M
    const size_t nQKV = (size_t)BATCH * NHEADS * SEQ * HDIM;  // 4M each

    __bf16* xb     = (__bf16*)ws;
    __bf16* wqkvb  = xb + nX;
    __bf16* wprojb = wqkvb + nWq;
    __bf16* Qb     = wprojb + nWp;
    __bf16* Kb     = Qb + nQKV;
    __bf16* Vb     = Kb + nQKV;
    __bf16* Ob     = Vb + nQKV;

    cvt_f32_bf16<<<2048, 256, 0, stream>>>(x, xb, (int)nX);
    cvt_f32_bf16<<<2048, 256, 0, stream>>>(wqkv, wqkvb, (int)nWq);
    cvt_f32_bf16<<<2048, 256, 0, stream>>>(wproj, wprojb, (int)nWp);

    qkv_gemm<<<dim3(MROWS / 128, QKVCOLS / 64), 128, 0, stream>>>(
        xb, wqkvb, Qb, Kb, Vb);

    attn_kernel<<<dim3(BATCH * NHEADS, SEQ / 64), 128, 0, stream>>>(
        Qb, Kb, Vb, Ob);

    proj_gemm<<<dim3(MROWS / 128, DIM / 64), 128, 0, stream>>>(
        Ob, wprojb, bproj, out);
}
